// EdgeDecoder_74388833566986
// MI455X (gfx1250) — matrix operand order repaired and checked by reading, not compiled, not run on hardware
//
#include <hip/hip_runtime.h>
#include <hip/hip_bf16.h>
#include <math.h>

typedef __attribute__((ext_vector_type(16))) __bf16 v16bf;
typedef __attribute__((ext_vector_type(8)))  __bf16 v8bf;
typedef __attribute__((ext_vector_type(8)))  float  v8f;
typedef __attribute__((ext_vector_type(4)))  unsigned int v4u;
typedef __attribute__((ext_vector_type(8)))  int v8i;
typedef __attribute__((ext_vector_type(4)))  int v4i;

#define KTOT   2304          // 1024 (scalar) + 1024 (vector) + 256 (tensor)
#define FEATF  208
#define RECU   216           // bf16 units per staged feature record (432 B)
#define ESTR   68            // estore row stride in floats

// ---- LDS layout (bytes) ----
#define OFF_FEAT 0
#define SZ_FEAT  (8 * 32 * RECU * 2)        // 110592
#define OFF_B    (OFF_FEAT + SZ_FEAT)
#define SZ_B     (512 * 48)                 // 24576 per buffer, double buffered
#define OFF_ES   (OFF_B + 2 * SZ_B)         // 159744
#define SZ_ES    (8 * 16 * ESTR * 4)        // 34816
#define OFF_MLP  (OFF_ES + SZ_ES)           // 194560
#define SZ_MLP   ((2048 + 32 + 32 + 16) * 4)
#define SMEM_BYTES (OFF_MLP + SZ_MLP)       // 203072 (< 320KB per WGP)

#if __has_builtin(__builtin_amdgcn_tensor_load_to_lds)
#define HAVE_TDM 1
#else
#define HAVE_TDM 0
#endif

#if __has_builtin(__builtin_amdgcn_s_wait_tensorcnt)
#define WAIT_TENSOR(n) __builtin_amdgcn_s_wait_tensorcnt(n)
#else
#define WAIT_TENSOR(n) asm volatile("s_wait_tensorcnt 0x0" ::: "memory")
#endif

#define SHUF16(lo, hi) __builtin_shufflevector((lo), (hi), 0,1,2,3,4,5,6,7,8,9,10,11,12,13,14,15)

// Pre-scale + transpose W into Wt[n][k] (bf16), k in fused 2304 ordering.
__global__ void prep_wt_kernel(const float* __restrict__ W0, const float* __restrict__ W1,
                               const float* __restrict__ W2, __bf16* __restrict__ Wt) {
  int idx = blockIdx.x * blockDim.x + threadIdx.x;
  if (idx >= 64 * KTOT) return;
  int n = idx / KTOT;
  int k = idx - n * KTOT;
  const float aS = 1.0f / 48.0f;                 // 1/sqrt(2304)
  const float aV = aS * 0.57735026919f;          // /sqrt(3)
  const float aT = aS * 0.44721359549f;          // /sqrt(5)
  float val;
  if (k < 1024) {
    int u = k >> 5, v = k & 31;
    val = W0[(u * 32 + v) * 64 + n] * aS;
  } else if (k < 2048) {
    int kk = k - 1024; int u = kk >> 5, v = kk & 31;
    val = W1[(u * 32 + v) * 64 + n] * aV;
  } else {
    int kk = k - 2048; int u = kk >> 4, v = kk & 15;
    val = W2[(u * 16 + v) * 64 + n] * aT;
  }
  Wt[(size_t)n * KTOT + k] = (__bf16)val;
}

#if HAVE_TDM
// One TDM descriptor loads a whole 24KB B chunk (4 k-steps x 64 cols) into LDS in
// fragment-slot order: dims = 32B row (x) , n(16), half(2), nt(4), ksl(4); pad 16B
// after every 32B row -> 48B slots.
__device__ __forceinline__ void tdm_issue_chunk(const __bf16* Wt, int chunk, int buf) {
  unsigned long long g = (unsigned long long)(uintptr_t)Wt + (unsigned long long)chunk * 256ull;
  v4u g0;
  g0[0] = 1u;                                        // count=1 (valid), user mode
  g0[1] = (unsigned)(OFF_B + buf * SZ_B);            // lds_addr
  g0[2] = (unsigned)g;                               // global_addr[31:0]
  g0[3] = (unsigned)((g >> 32) & 0x01FFFFFFull) | (2u << 30);  // addr[56:32] | type=2
  v8i g1;
  g1[0] = (int)((1u << 16)      // data_size = 2B
              | (1u << 20)      // pad_enable
              | (2u << 22)      // pad_interval: 8 dwords (32B)
              | (3u << 25));    // pad_amount: 4 dwords (16B)
  g1[1] = (int)(2304u << 16);                // tensor_dim0 = 2304 (lo16)
  g1[2] = (int)(64u << 16);                  // dim0 hi16=0 | tensor_dim1 = 64 (lo16)
  g1[3] = (int)(16u << 16);                  // dim1 hi16=0 | tile_dim0 = 16
  g1[4] = (int)(16u | (2u << 16));           // tile_dim1 = 16 | tile_dim2 = 2
  g1[5] = 2304;                              // tensor_dim0_stride (lo32)
  g1[6] = (int)(16u << 16);                  // stride0 hi16=0 | tensor_dim1_stride = 16
  g1[7] = 0;                                 // stride1 hi32
  v4i g2;
  g2[0] = 2;                                 // tensor_dim2
  g2[1] = 4;                                 // tensor_dim3
  g2[2] = 36864;                             // tensor_dim2_stride = 16*2304 (lo32)
  g2[3] = (int)(4u << 16);                   // stride2 hi16=0 | tile_dim3 = 4
  v4i g3;
  g3[0] = 32;                                // tensor_dim3_stride (lo32)
  g3[1] = (int)(4u << 16);                   // stride3 hi16=0 | tensor_dim4 = 4 (lo16)
  g3[2] = (int)(4u << 16);                   // dim4 hi16=0 | tile_dim4 = 4
  g3[3] = 0;
  v8i g4 = {0, 0, 0, 0, 0, 0, 0, 0};         // trailing group (6-arg toolchain form)
  __builtin_amdgcn_tensor_load_to_lds(g0, g1, g2, g3, g4, 0);
}
#endif

__global__ __launch_bounds__(256)
void edge_decoder_kernel(const float* __restrict__ nodes,
                         const int*   __restrict__ ei,
                         const __bf16* __restrict__ Wt,
                         const float* __restrict__ A1,
                         const float* __restrict__ b1,
                         const float* __restrict__ A2,
                         const float* __restrict__ b2,
                         float* __restrict__ out,
                         int nEdges)
{
  extern __shared__ char smem[];
  const int tid  = threadIdx.x;
  const int wave = tid >> 5;
  const int lane = tid & 31;
  const int r    = lane & 15;       // edge row (A-matrix M)
  const int half = lane >> 4;       // K sub-pattern half per ISA layout
  const int edgeBase = blockIdx.x * 128 + wave * 16;

  __bf16* featW = (__bf16*)(smem + OFF_FEAT) + wave * (32 * RECU);
  float*  esW   = (float*)(smem + OFF_ES) + wave * (16 * ESTR);
  float*  sA1   = (float*)(smem + OFF_MLP);
  float*  sb1   = sA1 + 2048;
  float*  sA2   = sb1 + 32;
  float*  sb2   = sA2 + 32;

  // ---- stage MLP constants ----
  for (int i = tid; i < 2048; i += 256) sA1[i] = A1[i];
  if (tid < 32) { sb1[tid] = b1[tid]; sA2[tid] = A2[tid]; }
  if (tid == 0) sb2[0] = b2[0];

  // ---- stage 16 edges' endpoint features (bf16, component-transposed) ----
  {
    int e = edgeBase + r; if (e > nEdges - 1) e = nEdges - 1;
    int nodeId = ei[(size_t)half * nEdges + e];
    const float* f = nodes + (size_t)nodeId * FEATF;
    __bf16* rec = featW + (half * 16 + r) * RECU;
    #pragma unroll
    for (int u = 0; u < 32; ++u) rec[u] = (__bf16)f[u];
    #pragma unroll
    for (int u = 0; u < 32; ++u) {
      rec[32 + u] = (__bf16)f[32 + 3 * u + 0];
      rec[64 + u] = (__bf16)f[32 + 3 * u + 1];
      rec[96 + u] = (__bf16)f[32 + 3 * u + 2];
    }
    #pragma unroll
    for (int u = 0; u < 16; ++u) {
      #pragma unroll
      for (int m = 0; m < 5; ++m)
        rec[128 + m * 16 + u] = (__bf16)f[128 + 5 * u + m];
    }
  }

  const __bf16* rec1 = featW + r * RECU;          // x1 of edge r
  const __bf16* rec2 = featW + (16 + r) * RECU;   // x2 of edge r
  const int vb = half * 16;

  v8f acc[4] = {};

  auto splat = [](__bf16 x) { v8bf v = {x, x, x, x, x, x, x, x}; return v; };

  auto wmma4 = [&](v16bf av, const char* bB, int ksl) {
    {
      const char* p = bB + (ksl * 128 + 0 * 32 + lane) * 48;
      v8bf lo = *(const v8bf*)p, hi = *(const v8bf*)(p + 16);
      acc[0] = __builtin_amdgcn_wmma_f32_16x16x32_bf16(false, av, false, SHUF16(lo, hi), (short)0, acc[0], false, false);
    }
    {
      const char* p = bB + (ksl * 128 + 1 * 32 + lane) * 48;
      v8bf lo = *(const v8bf*)p, hi = *(const v8bf*)(p + 16);
      acc[1] = __builtin_amdgcn_wmma_f32_16x16x32_bf16(false, av, false, SHUF16(lo, hi), (short)0, acc[1], false, false);
    }
    {
      const char* p = bB + (ksl * 128 + 2 * 32 + lane) * 48;
      v8bf lo = *(const v8bf*)p, hi = *(const v8bf*)(p + 16);
      acc[2] = __builtin_amdgcn_wmma_f32_16x16x32_bf16(false, av, false, SHUF16(lo, hi), (short)0, acc[2], false, false);
    }
    {
      const char* p = bB + (ksl * 128 + 3 * 32 + lane) * 48;
      v8bf lo = *(const v8bf*)p, hi = *(const v8bf*)(p + 16);
      acc[3] = __builtin_amdgcn_wmma_f32_16x16x32_bf16(false, av, false, SHUF16(lo, hi), (short)0, acc[3], false, false);
    }
  };

#if !HAVE_TDM
  auto fillCoop = [&](int chunk, int b) {
    char* base = smem + OFF_B + b * SZ_B;
    #pragma unroll
    for (int ss = 0; ss < 2; ++ss) {
      int s = tid + ss * 256;
      int ksl2 = s >> 7, rem = s & 127, nt = rem >> 5, ln = rem & 31;
      int n = (ln & 15) + 16 * nt;
      int kst = chunk * 128 + ksl2 * 32 + 16 * (ln >> 4);
      const __bf16* gp = Wt + (size_t)n * KTOT + kst;
      char* dst = base + s * 48;
      *(v8bf*)dst        = *(const v8bf*)gp;
      *(v8bf*)(dst + 16) = *(const v8bf*)(gp + 8);
    }
  };
#endif

#if HAVE_TDM
#define PRE(c)  do { if (tid < 32) { if ((c) < 17) WAIT_TENSOR(1); else WAIT_TENSOR(0); } } while (0)
#define POST(c) do { if (tid < 32 && (c) + 2 < 18) tdm_issue_chunk(Wt, (c) + 2, bufi); } while (0)
#else
#define PRE(c)  do {} while (0)
#define POST(c) do { if ((c) + 2 < 18) fillCoop((c) + 2, bufi); } while (0)
#endif

  int bufi = 0;
#if HAVE_TDM
  if (tid < 32) { tdm_issue_chunk(Wt, 0, 0); tdm_issue_chunk(Wt, 1, 1); }
#else
  fillCoop(0, 0);
  fillCoop(1, 1);
#endif

  // ================= Phase S: chunks 0..7 (scalar x scalar) =================
  {
    v8bf sa = *(const v8bf*)(rec2 + vb);
    v8bf sb = *(const v8bf*)(rec2 + 8 + vb);
    for (int c = 0; c < 8; ++c) {
      PRE(c);
      __syncthreads();
      const char* bB = smem + OFF_B + bufi * SZ_B;
      #pragma unroll
      for (int ksl = 0; ksl < 4; ++ksl) {
        int u = c * 4 + ksl;
        v8bf sv = splat(rec1[u]);
        v8bf lo = sv * sa;
        v8bf hi = sv * sb;
        wmma4(SHUF16(lo, hi), bB, ksl);
      }
      __syncthreads();
      POST(c);
      bufi ^= 1;
    }
  }
  // ================= Phase V: chunks 8..15 (vector dot3) =================
  {
    v8bf b0a = *(const v8bf*)(rec2 + 32 + vb);
    v8bf b1a = *(const v8bf*)(rec2 + 64 + vb);
    v8bf b2a = *(const v8bf*)(rec2 + 96 + vb);
    v8bf b0b = *(const v8bf*)(rec2 + 40 + vb);
    v8bf b1b = *(const v8bf*)(rec2 + 72 + vb);
    v8bf b2b = *(const v8bf*)(rec2 + 104 + vb);
    for (int c = 8; c < 16; ++c) {
      PRE(c);
      __syncthreads();
      const char* bB = smem + OFF_B + bufi * SZ_B;
      #pragma unroll
      for (int ksl = 0; ksl < 4; ++ksl) {
        int u = c * 4 + ksl - 32;
        v8bf A0 = splat(rec1[32 + u]);
        v8bf A1v = splat(rec1[64 + u]);
        v8bf A2v = splat(rec1[96 + u]);
        v8bf lo = A0 * b0a + A1v * b1a + A2v * b2a;
        v8bf hi = A0 * b0b + A1v * b1b + A2v * b2b;
        wmma4(SHUF16(lo, hi), bB, ksl);
      }
      __syncthreads();
      POST(c);
      bufi ^= 1;
    }
  }
  // ================= Phase T: chunks 16..17 (tensor dot5) =================
  {
    v8bf t20 = *(const v8bf*)(rec2 + 128 +  0);
    v8bf t21 = *(const v8bf*)(rec2 + 128 + 16);
    v8bf t22 = *(const v8bf*)(rec2 + 128 + 32);
    v8bf t23 = *(const v8bf*)(rec2 + 128 + 48);
    v8bf t24 = *(const v8bf*)(rec2 + 128 + 64);
    v8bf t20h = *(const v8bf*)(rec2 + 128 +  8);
    v8bf t21h = *(const v8bf*)(rec2 + 128 + 24);
    v8bf t22h = *(const v8bf*)(rec2 + 128 + 40);
    v8bf t23h = *(const v8bf*)(rec2 + 128 + 56);
    v8bf t24h = *(const v8bf*)(rec2 + 128 + 72);
    for (int c = 16; c < 18; ++c) {
      PRE(c);
      __syncthreads();
      const char* bB = smem + OFF_B + bufi * SZ_B;
      #pragma unroll
      for (int ksl = 0; ksl < 4; ++ksl) {
        int q = c * 4 + ksl - 64;
        v8bf lo = splat(rec1[128 +  0 + 2 * q + half]) * t20
                + splat(rec1[128 + 16 + 2 * q + half]) * t21
                + splat(rec1[128 + 32 + 2 * q + half]) * t22
                + splat(rec1[128 + 48 + 2 * q + half]) * t23
                + splat(rec1[128 + 64 + 2 * q + half]) * t24;
        v8bf hi = splat(rec1[128 +  0 + 2 * q + half]) * t20h
                + splat(rec1[128 + 16 + 2 * q + half]) * t21h
                + splat(rec1[128 + 32 + 2 * q + half]) * t22h
                + splat(rec1[128 + 48 + 2 * q + half]) * t23h
                + splat(rec1[128 + 64 + 2 * q + half]) * t24h;
        wmma4(SHUF16(lo, hi), bB, ksl);
      }
      __syncthreads();
      POST(c);
      bufi ^= 1;
    }
  }

  // ---- scatter C (VGPR i -> M=i+8*half, N=lane%16) to LDS ----
  const int nb = lane & 15;
  #pragma unroll
  for (int i = 0; i < 8; ++i) {
    int M = i + 8 * half;
    esW[M * ESTR + nb +  0] = acc[0][i];
    esW[M * ESTR + nb + 16] = acc[1][i];
    esW[M * ESTR + nb + 32] = acc[2][i];
    esW[M * ESTR + nb + 48] = acc[3][i];
  }

  // ---- fused MLP epilogue, both lane halves active ----
  // half h accumulates over w in [32h, 32h+32); partials combined via LDS (same wave,
  // DS ops are in-order per wave so no barrier needed).
  {
    float hj[32];
    #pragma unroll
    for (int j = 0; j < 32; ++j) hj[j] = half ? 0.0f : sb1[j];
    const float* er = esW + r * ESTR + 32 * half;
    const float* a1base = sA1 + (32 * half) * 32;
    #pragma unroll 4
    for (int w = 0; w < 32; ++w) {
      float ew = er[w];
      const float* a1r = a1base + w * 32;
      #pragma unroll
      for (int j = 0; j < 32; ++j) hj[j] += ew * a1r[j];
    }
    float* comb = (float*)(smem + OFF_B) + (wave * 16 + r) * 34;  // reuse free B buffer
    if (half) {
      #pragma unroll
      for (int j = 0; j < 32; ++j) comb[j] = hj[j];
    }
    if (!half) {
      int e = edgeBase + r;
      if (e < nEdges) {
        float o = sb2[0];
        #pragma unroll
        for (int j = 0; j < 32; ++j) {
          float x = hj[j] + comb[j];
          float s = x / (1.0f + __expf(-x));
          o += s * sA2[j];
        }
        out[e] = o;
      }
    }
  }
#undef PRE
#undef POST
}

extern "C" void kernel_launch(void* const* d_in, const int* in_sizes, int n_in,
                              void* d_out, int out_size, void* d_ws, size_t ws_size,
                              hipStream_t stream) {
  const float* nodes = (const float*)d_in[0];
  const int*   ei    = (const int*)  d_in[1];
  const float* W0    = (const float*)d_in[2];
  const float* W1    = (const float*)d_in[3];
  const float* W2    = (const float*)d_in[4];
  const float* A1    = (const float*)d_in[5];
  const float* b1    = (const float*)d_in[6];
  const float* A2    = (const float*)d_in[7];
  const float* b2    = (const float*)d_in[8];
  const int nEdges = in_sizes[1] / 2;

  __bf16* Wt = (__bf16*)d_ws;   // 64*2304*2 = 294912 bytes of scratch

  {
    int total = 64 * KTOT;
    prep_wt_kernel<<<(total + 255) / 256, 256, 0, stream>>>(W0, W1, W2, Wt);
  }

  (void)hipFuncSetAttribute((const void*)edge_decoder_kernel,
                            hipFuncAttributeMaxDynamicSharedMemorySize, SMEM_BYTES);
  int grid = (nEdges + 127) / 128;   // 128 edges per 256-thread block (8 waves x 16)
  edge_decoder_kernel<<<grid, 256, SMEM_BYTES, stream>>>(
      nodes, ei, Wt, A1, b1, A2, b2, (float*)d_out, nEdges);
}
